// HierarchicalGenerator_12610023981142
// MI455X (gfx1250) — matrix operand order repaired and checked by reading, not compiled, not run on hardware
//
#include <hip/hip_runtime.h>
#include <math.h>

// ---------------- problem constants ----------------
#define BB   64
#define TT   512
#define HH   768
#define ZS   256
#define ZT   128
#define G3   2304            // 3*H
#define GK0  896             // ZT + H
#define NPOS (BB*TT)         // 32768
#define NHEAD 32             // padded head count (17 real)

// output layout (flat float32, concatenated in reference return order)
constexpr size_t O_SCONT  = 0;                       // [64,4]
constexpr size_t O_DIAG   = O_SCONT  + 64*4;         // [64,10]
constexpr size_t O_EMB    = O_DIAG   + 64*10;        // [64,16]
constexpr size_t O_TCONT  = O_EMB    + 64*16;        // [64,512,8]
constexpr size_t O_DEATH  = O_TCONT  + (size_t)NPOS*8;   // [64,512,2]
constexpr size_t O_STAGE  = O_DEATH  + (size_t)NPOS*2;   // [64,512,6]
constexpr size_t O_VMASK  = O_STAGE  + (size_t)NPOS*6;   // [64,512,1]
constexpr size_t O_VTIME  = O_VMASK  + (size_t)NPOS;     // [64,512]

// ---------------- WMMA types ----------------
typedef __attribute__((ext_vector_type(16))) _Float16 v16h;
typedef __attribute__((ext_vector_type(8)))  float    v8f;

// ---------------- small helpers ----------------
__device__ __forceinline__ float sigm(float x) { return 1.f / (1.f + __expf(-x)); }
__device__ __forceinline__ float softplus(float x) {
  if (x > 20.f) return x;
  if (x < -20.f) return __expf(x);
  return log1pf(__expf(x));
}
__device__ __forceinline__ unsigned hash3(unsigned a, unsigned b, unsigned c) {
  unsigned h = a * 0x9E3779B9u ^ b * 0x85EBCA6Bu ^ c * 0xC2B2AE35u;
  h ^= h >> 16; h *= 0x7FEB352Du; h ^= h >> 15; h *= 0x846CA68Bu; h ^= h >> 16;
  return h;
}
__device__ __forceinline__ float uni01(unsigned a, unsigned b, unsigned c) {
  return ((hash3(a, b, c) >> 8) + 0.5f) * (1.0f / 16777216.0f);
}
__device__ __forceinline__ float gumbel(float u) { return -__logf(-__logf(u)); }

// async global -> LDS 16-byte copy (CDNA5 async path, ISA §15.18.3 op 98).
// lds_off is the byte offset into the workgroup LDS allocation (VDST VGPR).
// lds_base (generic pointer to the LDS array) is passed as an UNUSED asm input so the
// LDS global's address escapes into a memory-clobbering asm: this forces LLVM to treat
// the array as written here (otherwise "never-stored" analysis folds later reads).
__device__ __forceinline__ void async_g2l_b128(const void* lds_base, unsigned lds_off,
                                               const void* gsrc) {
  asm volatile("global_load_async_to_lds_b128 %0, %1, off"
               :: "v"(lds_off), "v"(gsrc), "v"(lds_base) : "memory");
}
__device__ __forceinline__ void wait_async0() {
  asm volatile("s_wait_asynccnt 0" ::: "memory");
}

// load A fragment (16x32 f16, M x K) from row-major src.
// lane 0-15: M=lane, K 0..15; lanes 16-31: same M rows, K 16..31.
template <typename PT>
__device__ __forceinline__ v16h load_a_frag(const PT* row, int k0, int g) {
  v16h a;
#pragma unroll
  for (int j = 0; j < 16; j += 2) {
    int k = k0 + g * 16 + j;
    a[j]     = (_Float16)row[k];
    a[j + 1] = (_Float16)row[k + 1];
  }
  return a;
}
// load B fragment (32x16 f16, K x N) where B(k,n) = W[n*ld + k] (W row-major [N,K]).
// lane n = lane&15, lanes 16-31 take K+16. 16 contiguous halves -> b128-pairs.
template <typename PT>
__device__ __forceinline__ v16h load_b_frag(const PT* wrow /* = W + n*ld + k0 + g*16 */) {
  v16h b;
#pragma unroll
  for (int j = 0; j < 16; ++j) b[j] = (_Float16)wrow[j];
  return b;
}

// ---------------- generic fp32 -> f16 convert ----------------
__global__ void cvt_f16_kernel(const float* __restrict__ in, _Float16* __restrict__ out, size_t n) {
  size_t i = (size_t)blockIdx.x * blockDim.x + threadIdx.x;
  if (i < n) out[i] = (_Float16)in[i];
}

// ---------------- small dense (static branch) ----------------
__global__ void dense_kernel(const float* __restrict__ X, const float* __restrict__ W,
                             const float* __restrict__ bias, float* __restrict__ out,
                             int M, int N, int K, int relu) {
  int idx = blockIdx.x * blockDim.x + threadIdx.x;
  if (idx >= M * N) return;
  int m = idx / N, n = idx % N;
  const float* x = X + (size_t)m * K;
  const float* w = W + (size_t)n * K;
  float s = bias[n];
  for (int k = 0; k < K; ++k) s = fmaf(x[k], w[k], s);
  if (relu) s = fmaxf(s, 0.f);
  out[idx] = s;
}

// ---------------- static heads: scont, diag (gumbel one-hot), emb ----------------
__global__ void static_heads_kernel(const float* __restrict__ sh,
                                    const float* __restrict__ scontW, const float* __restrict__ scontB,
                                    const float* __restrict__ diagW,  const float* __restrict__ diagB,
                                    const float* __restrict__ embW,   const float* __restrict__ embB,
                                    float* __restrict__ out) {
  int b = blockIdx.x;
  int tid = threadIdx.x;       // 64 threads
  __shared__ float dl[10];
  const float* h = sh + (size_t)b * HH;
  if (tid < 30) {
    const float* w; float s;
    if (tid < 4)        { w = scontW + tid * HH;        s = scontB[tid]; }
    else if (tid < 14)  { w = diagW + (tid - 4) * HH;   s = diagB[tid - 4]; }
    else                { w = embW  + (tid - 14) * HH;  s = embB[tid - 14]; }
    for (int k = 0; k < HH; ++k) s = fmaf(h[k], w[k], s);
    if (tid < 4)       out[O_SCONT + b * 4 + tid] = s;
    else if (tid < 14) dl[tid - 4] = s;
    else               out[O_EMB + b * 16 + (tid - 14)] = s;
  }
  __syncthreads();
  if (tid == 0) {
    int best = 0; float bv = -1e30f;
    for (int j = 0; j < 10; ++j) {
      float v = dl[j] + gumbel(uni01(0xD1A6u, (unsigned)b, (unsigned)j));
      if (v > bv) { bv = v; best = j; }
    }
    for (int j = 0; j < 10; ++j) out[O_DIAG + b * 10 + j] = (j == best) ? 1.f : 0.f;
  }
}

// ---------------- build GRU input [B*T, 896] f16 = [z_temporal ; static_h] ----------------
__global__ void build_gin_kernel(const float* __restrict__ zt, const float* __restrict__ sh,
                                 _Float16* __restrict__ gin) {
  size_t idx = (size_t)blockIdx.x * blockDim.x + threadIdx.x;
  const size_t total = (size_t)NPOS * GK0;
  if (idx >= total) return;
  int c = (int)(idx % GK0);
  size_t row = idx / GK0;          // row = b*T + t
  int b = (int)(row / TT);
  float v = (c < ZT) ? zt[row * ZT + c] : sh[(size_t)b * HH + (c - ZT)];
  gin[idx] = (_Float16)v;
}

// ---------------- pack 17 temporal-head vectors into padded [32,768] f16 ----------------
__global__ void pack_heads_kernel(const float* __restrict__ timeW, const float* __restrict__ visitW,
                                  const float* __restrict__ deathW, const float* __restrict__ tcontW,
                                  const float* __restrict__ stageW,
                                  const float* __restrict__ timeB, const float* __restrict__ visitB,
                                  const float* __restrict__ deathB, const float* __restrict__ tcontB,
                                  const float* __restrict__ stageB,
                                  _Float16* __restrict__ Wpk, float* __restrict__ bpk) {
  int idx = blockIdx.x * blockDim.x + threadIdx.x;
  if (idx < NHEAD * HH) {
    int o = idx / HH, k = idx % HH;
    float v = 0.f;
    if (o == 0)       v = timeW[k];
    else if (o == 1)  v = visitW[k];
    else if (o == 2)  v = deathW[k];
    else if (o < 11)  v = tcontW[(o - 3) * HH + k];
    else if (o < 17)  v = stageW[(o - 11) * HH + k];
    Wpk[idx] = (_Float16)v;
  }
  if (idx < NHEAD) {
    float v = 0.f;
    if (idx == 0)      v = timeB[0];
    else if (idx == 1) v = visitB[0];
    else if (idx == 2) v = deathB[0];
    else if (idx < 11) v = tcontB[idx - 3];
    else if (idx < 17) v = stageB[idx - 11];
    bpk[idx] = v;
  }
}

// ---------------- big GEMM via WMMA: out[M,N] = X[M,K] * W[N,K]^T + bias ----------------
// block = 128 threads (4 waves); all waves share one 16-col B tile, async-staged in LDS.
// wave w -> rows [blockIdx.x*64 + 16w, +16), cols [blockIdx.y*16, +16)
__global__ void __launch_bounds__(128)
gemm_wmma_kernel(const _Float16* __restrict__ X, const _Float16* __restrict__ W,
                 const float* __restrict__ bias, float* __restrict__ out,
                 int M, int N, int K) {
  __shared__ _Float16 ldsW[16 * GK0];       // 28 KB max (K<=896): 16 weight rows
  int tid = threadIdx.x;
  int wave = tid >> 5;
  int lane = tid & 31;
  int g = lane >> 4, lm = lane & 15;
  int m0 = blockIdx.x * 64 + wave * 16;
  int n0 = blockIdx.y * 16;

  // async-stage the contiguous 16*K half block W[n0..n0+15][*] into LDS
  {
    const _Float16* src = W + (size_t)n0 * K;
    int chunks = (16 * K) >> 3;             // 16-byte chunks
    for (int c = tid; c < chunks; c += 128)
      async_g2l_b128((const void*)ldsW, (unsigned)(c << 4), src + (c << 3));
    wait_async0();
  }
  __syncthreads();

  const _Float16* xrow = X + (size_t)(m0 + lm) * K;
  const _Float16* wrow = ldsW + lm * K + g * 16;
  v8f acc = {0.f, 0.f, 0.f, 0.f, 0.f, 0.f, 0.f, 0.f};
  for (int k0 = 0; k0 < K; k0 += 32) {
    __builtin_prefetch(xrow + k0 + 256, 0, 1);   // global_prefetch_b8: activation stream
    v16h a = load_a_frag(xrow, k0, g);
    v16h b = load_b_frag(wrow + k0);
    acc = __builtin_amdgcn_wmma_f32_16x16x32_f16(false, a, false, b, (short)0, acc,
                                                 false, false);
  }
  float bv = bias[n0 + lm];
#pragma unroll
  for (int r = 0; r < 8; ++r)
    out[(size_t)(m0 + r + 8 * g) * N + n0 + lm] = acc[r] + bv;
}

// ---------------- persistent GRU scan: 1 block = 16 batch rows, h lives in LDS ----------------
// xg: [B*T, 2304] fp32 (bih folded in). Whh: [2304,768] f16 (L2-resident), bhh fp32.
// hseq out: [B*T, 768] f16.
__global__ void __launch_bounds__(256)
gru_scan_kernel(const float* __restrict__ xg, const _Float16* __restrict__ Whh,
                const float* __restrict__ bhh, _Float16* __restrict__ hseq) {
  __shared__ _Float16 hH[16 * HH];          // 24 KB: current hidden state, f16
  int tid = threadIdx.x;
  int wave = tid >> 5, lane = tid & 31;
  int g = lane >> 4, lm = lane & 15;
  int b0 = blockIdx.x * 16;
  for (int i = tid; i < 16 * HH; i += 256) hH[i] = (_Float16)0.f;
  __syncthreads();

  for (int t = 0; t < TT; ++t) {
    float hnew[6][8];
#pragma unroll 1
    for (int i = 0; i < 6; ++i) {
      int col0 = (wave * 6 + i) * 16;       // 0..752, per-gate column tile
      const v8f zero = {0.f, 0.f, 0.f, 0.f, 0.f, 0.f, 0.f, 0.f};
      v8f ar = zero, az = zero, an = zero;
      const _Float16* hrow = &hH[lm * HH];
      const _Float16* wr = Whh + (size_t)(0 * HH + col0 + lm) * HH + g * 16;
      const _Float16* wz = Whh + (size_t)(1 * HH + col0 + lm) * HH + g * 16;
      const _Float16* wn = Whh + (size_t)(2 * HH + col0 + lm) * HH + g * 16;
      for (int k0 = 0; k0 < HH; k0 += 32) {
        __builtin_prefetch(wr + k0 + 256, 0, 1);
        v16h a = load_a_frag(hrow, k0, g);
        v16h br = load_b_frag(wr + k0);
        v16h bz = load_b_frag(wz + k0);
        v16h bn = load_b_frag(wn + k0);
        ar = __builtin_amdgcn_wmma_f32_16x16x32_f16(false, a, false, br, (short)0, ar, false, false);
        az = __builtin_amdgcn_wmma_f32_16x16x32_f16(false, a, false, bz, (short)0, az, false, false);
        an = __builtin_amdgcn_wmma_f32_16x16x32_f16(false, a, false, bn, (short)0, an, false, false);
      }
      // elementwise GRU update: same lane/slot holds matching r/z/n columns
      int col = col0 + lm;
      float bhr = bhh[col], bhz = bhh[HH + col], bhn = bhh[2 * HH + col];
#pragma unroll
      for (int r = 0; r < 8; ++r) {
        int mloc = r + 8 * g;
        const float* x = xg + ((size_t)(b0 + mloc) * TT + t) * G3;
        float rr = sigm(x[col] + ar[r] + bhr);
        float zz = sigm(x[HH + col] + az[r] + bhz);
        float nn = tanhf(x[2 * HH + col] + rr * (an[r] + bhn));
        float hold = (float)hH[mloc * HH + col];
        hnew[i][r] = (1.f - zz) * nn + zz * hold;
      }
    }
    __syncthreads();                         // everyone done reading old h
#pragma unroll
    for (int i = 0; i < 6; ++i) {
      int col = (wave * 6 + i) * 16 + lm;
#pragma unroll
      for (int r = 0; r < 8; ++r) {
        int mloc = r + 8 * g;
        _Float16 hv = (_Float16)hnew[i][r];
        hH[mloc * HH + col] = hv;
        hseq[((size_t)(b0 + mloc) * TT + t) * HH + col] = hv;
      }
    }
    __syncthreads();                         // new h visible before next step
  }
}

// ---------------- head post-process: split packed [B*T,32] head GEMM output ----------------
__global__ void head_post_kernel(const float* __restrict__ hd,
                                 float* __restrict__ dt, float* __restrict__ vp,
                                 float* __restrict__ hz, float* __restrict__ tc,
                                 float* __restrict__ st) {
  int row = blockIdx.x * blockDim.x + threadIdx.x;
  if (row >= NPOS) return;
  const float* v = hd + (size_t)row * NHEAD;
  dt[row] = softplus(v[0]);
  vp[row] = sigm(v[1]);
  hz[row] = sigm(v[2]);
#pragma unroll
  for (int o = 0; o < 8; ++o) tc[(size_t)row * 8 + o] = v[3 + o];
#pragma unroll
  for (int o = 0; o < 6; ++o) st[(size_t)row * 6 + o] = v[11 + o];
}

// ---------------- per-batch-row sequential scans + masked outputs ----------------
__global__ void __launch_bounds__(256)
finalize_kernel(const float* __restrict__ dt, const float* __restrict__ vp,
                const float* __restrict__ hzv, const float* __restrict__ tc,
                const float* __restrict__ st, float* __restrict__ out) {
  int b = blockIdx.x;
  int tid = threadIdx.x;
  __shared__ float cum[TT], mask[TT], irr[TT];
  __shared__ float vmax_s;
  if (tid == 0) {
    float run = 0.f, vmax = -1e30f;
    for (int t = 0; t < TT; ++t) {
      run += dt[(size_t)b * TT + t];
      cum[t] = run;
      vmax = fmaxf(vmax, run);
    }
    vmax_s = vmax;
  } else if (tid == 32) {
    float m = 1.f;
    for (int t = 0; t < TT; ++t) {
      float u = uni01(0x5151u, (unsigned)b, (unsigned)t);
      float s = (t == 0) ? 1.f : ((u < vp[(size_t)b * TT + t]) ? 1.f : 0.f);
      m *= s;
      mask[t] = m;
    }
  } else if (tid == 64) {
    float prev = 0.f;
    for (int t = 0; t < TT; ++t) {
      float u = uni01(0xDEADu, (unsigned)b, (unsigned)t);
      float flip = (u < hzv[(size_t)b * TT + t] * (1.f - prev)) ? 1.f : 0.f;
      prev = fminf(prev + flip, 1.f);
      irr[t] = prev;
    }
  }
  __syncthreads();
  float vmax = vmax_s;
  for (int t = tid; t < TT; t += 256) {
    size_t r = (size_t)b * TT + t;
    float m = mask[t];
    out[O_VTIME + r] = cum[t] / (vmax + 1e-8f);
    out[O_VMASK + r] = m;
#pragma unroll
    for (int o = 0; o < 8; ++o) out[O_TCONT + r * 8 + o] = tc[r * 8 + o] * m;
    float ir = irr[t];
    out[O_DEATH + r * 2 + 0] = (1.f - ir) * m;
    out[O_DEATH + r * 2 + 1] = ir * m;
    int best = 0; float bv = -1e30f;
#pragma unroll
    for (int o = 0; o < 6; ++o) {
      float v = st[r * 6 + o] + gumbel(uni01(0x57A6u, (unsigned)r, (unsigned)o));
      if (v > bv) { bv = v; best = o; }
    }
#pragma unroll
    for (int o = 0; o < 6; ++o) out[O_STAGE + r * 6 + o] = (o == best) ? m : 0.f;
  }
}

// ---------------- launch ----------------
extern "C" void kernel_launch(void* const* d_in, const int* in_sizes, int n_in,
                              void* d_out, int out_size, void* d_ws, size_t ws_size,
                              hipStream_t stream) {
  (void)in_sizes; (void)n_in; (void)out_size; (void)ws_size;
  const float* z_static  = (const float*)d_in[0];
  const float* z_temporal= (const float*)d_in[1];
  const float* fcs_W1 = (const float*)d_in[2];  const float* fcs_b1 = (const float*)d_in[3];
  const float* fcs_W2 = (const float*)d_in[4];  const float* fcs_b2 = (const float*)d_in[5];
  const float* Wih0 = (const float*)d_in[6];    const float* Whh0 = (const float*)d_in[7];
  const float* bih0 = (const float*)d_in[8];    const float* bhh0 = (const float*)d_in[9];
  const float* Wih1 = (const float*)d_in[10];   const float* Whh1 = (const float*)d_in[11];
  const float* bih1 = (const float*)d_in[12];   const float* bhh1 = (const float*)d_in[13];
  const float* timeW = (const float*)d_in[14];  const float* timeB = (const float*)d_in[15];
  const float* visitW= (const float*)d_in[16];  const float* visitB= (const float*)d_in[17];
  const float* tcontW= (const float*)d_in[18];  const float* tcontB= (const float*)d_in[19];
  const float* deathW= (const float*)d_in[20];  const float* deathB= (const float*)d_in[21];
  const float* stageW= (const float*)d_in[22];  const float* stageB= (const float*)d_in[23];
  const float* scontW= (const float*)d_in[24];  const float* scontB= (const float*)d_in[25];
  const float* diagW = (const float*)d_in[26];  const float* diagB = (const float*)d_in[27];
  const float* embW  = (const float*)d_in[28];  const float* embB  = (const float*)d_in[29];
  float* out = (float*)d_out;

  // workspace carve-out (~490 MB assumed available)
  char* w = (char*)d_ws;
  auto carve = [&](size_t bytes) { void* p = (void*)w; w += (bytes + 255) & ~(size_t)255; return p; };
  float*     xg     = (float*)    carve((size_t)NPOS * G3 * 4);   // reused by both layers
  _Float16*  gin    = (_Float16*) carve((size_t)NPOS * GK0 * 2);
  _Float16*  h1     = (_Float16*) carve((size_t)NPOS * HH * 2);
  _Float16*  hsq    = (_Float16*) carve((size_t)NPOS * HH * 2);
  _Float16*  Wih0h  = (_Float16*) carve((size_t)G3 * GK0 * 2);
  _Float16*  Whh0h  = (_Float16*) carve((size_t)G3 * HH * 2);
  _Float16*  Wih1h  = (_Float16*) carve((size_t)G3 * HH * 2);
  _Float16*  Whh1h  = (_Float16*) carve((size_t)G3 * HH * 2);
  _Float16*  Wpk    = (_Float16*) carve((size_t)NHEAD * HH * 2);
  float*     bpk    = (float*)    carve((size_t)NHEAD * 4);
  float*     s1     = (float*)    carve((size_t)BB * HH * 4);
  float*     sh     = (float*)    carve((size_t)BB * HH * 4);
  float*     hd     = (float*)    carve((size_t)NPOS * NHEAD * 4);
  float*     dtb    = (float*)    carve((size_t)NPOS * 4);
  float*     vpb    = (float*)    carve((size_t)NPOS * 4);
  float*     hzb    = (float*)    carve((size_t)NPOS * 4);
  float*     tcb    = (float*)    carve((size_t)NPOS * 8 * 4);
  float*     stb    = (float*)    carve((size_t)NPOS * 6 * 4);

  auto cvt = [&](const float* src, _Float16* dst, size_t n) {
    cvt_f16_kernel<<<dim3((unsigned)((n + 255) / 256)), dim3(256), 0, stream>>>(src, dst, n);
  };
  cvt(Wih0, Wih0h, (size_t)G3 * GK0);
  cvt(Whh0, Whh0h, (size_t)G3 * HH);
  cvt(Wih1, Wih1h, (size_t)G3 * HH);
  cvt(Whh1, Whh1h, (size_t)G3 * HH);

  // static branch
  dense_kernel<<<dim3((BB * HH + 255) / 256), dim3(256), 0, stream>>>(z_static, fcs_W1, fcs_b1, s1, BB, HH, ZS, 1);
  dense_kernel<<<dim3((BB * HH + 255) / 256), dim3(256), 0, stream>>>(s1, fcs_W2, fcs_b2, sh, BB, HH, HH, 0);
  static_heads_kernel<<<dim3(BB), dim3(64), 0, stream>>>(sh, scontW, scontB, diagW, diagB, embW, embB, out);

  // GRU input concat -> f16; pack temporal-head weights
  {
    size_t total = (size_t)NPOS * GK0;
    build_gin_kernel<<<dim3((unsigned)((total + 255) / 256)), dim3(256), 0, stream>>>(z_temporal, sh, gin);
  }
  pack_heads_kernel<<<dim3((NHEAD * HH + 255) / 256), dim3(256), 0, stream>>>(
      timeW, visitW, deathW, tcontW, stageW, timeB, visitB, deathB, tcontB, stageB, Wpk, bpk);

  // layer 0: gate precompute (WMMA GEMM, async-staged weights), then persistent scan
  gemm_wmma_kernel<<<dim3(NPOS / 64, G3 / 16), dim3(128), 0, stream>>>(gin, Wih0h, bih0, xg, NPOS, G3, GK0);
  gru_scan_kernel<<<dim3(BB / 16), dim3(256), 0, stream>>>(xg, Whh0h, bhh0, h1);

  // layer 1: reuse xg buffer
  gemm_wmma_kernel<<<dim3(NPOS / 64, G3 / 16), dim3(128), 0, stream>>>(h1, Wih1h, bih1, xg, NPOS, G3, HH);
  gru_scan_kernel<<<dim3(BB / 16), dim3(256), 0, stream>>>(xg, Whh1h, bhh1, hsq);

  // temporal heads via WMMA GEMM (N=32 padded), then split + scans
  gemm_wmma_kernel<<<dim3(NPOS / 64, NHEAD / 16), dim3(128), 0, stream>>>(hsq, Wpk, bpk, hd, NPOS, NHEAD, HH);
  head_post_kernel<<<dim3((NPOS + 255) / 256), dim3(256), 0, stream>>>(hd, dtb, vpb, hzb, tcb, stb);
  finalize_kernel<<<dim3(BB), dim3(256), 0, stream>>>(dtb, vpb, hzb, tcb, stb, out);
}
